// LongRangeAttention_59365037965943
// MI455X (gfx1250) — matrix operand order repaired and checked by reading, not compiled, not run on hardware
//
#include <hip/hip_runtime.h>

// ---------------------------------------------------------------------------
// LongRangeAttention for MI455X (gfx1250, wave32, WMMA f32_16x16x32_f16)
// B=2, S=2048, D=1024, H=16, DK=64, chunk=64, MRP=128
// ---------------------------------------------------------------------------

typedef _Float16 half8  __attribute__((ext_vector_type(8)));
typedef _Float16 half16 __attribute__((ext_vector_type(16)));
typedef float    f32x8  __attribute__((ext_vector_type(8)));

#define NB   2
#define SEQ  2048
#define DIM  1024
#define NH   16
#define HDK  64
#define CHK  64
#define MRPC 128
#define MROW (NB * SEQ)   /* 4096 */

__device__ __forceinline__ half8 ld8(const _Float16* p) {
    return *reinterpret_cast<const half8*>(p);
}

__device__ __forceinline__ half16 cat16(half8 lo, half8 hi) {
    half16 r;
#pragma unroll
    for (int i = 0; i < 8; ++i) { r[i] = lo[i]; r[8 + i] = hi[i]; }
    return r;
}

__device__ __forceinline__ f32x8 wmma_f16(half16 a, half16 b, f32x8 c) {
    // 8 args: (neg_a, A, neg_b, B, c_mod, C, reuse_a, reuse_b)
    return __builtin_amdgcn_wmma_f32_16x16x32_f16(false, a, false, b,
                                                  (short)0, c, false, false);
}

__device__ __forceinline__ float rmax16(float v) {
    v = fmaxf(v, __shfl_xor(v, 1));
    v = fmaxf(v, __shfl_xor(v, 2));
    v = fmaxf(v, __shfl_xor(v, 4));
    v = fmaxf(v, __shfl_xor(v, 8));
    return v;
}

__device__ __forceinline__ float rsum16(float v) {
    v += __shfl_xor(v, 1);
    v += __shfl_xor(v, 2);
    v += __shfl_xor(v, 4);
    v += __shfl_xor(v, 8);
    return v;
}

// --- CDNA5 async copy: global -> LDS, tracked by ASYNCcnt -------------------
__device__ __forceinline__ uint32_t lds_off(const void* p) {
    // low 32 bits of a generic LDS pointer == wave-relative LDS byte address
    return (uint32_t)(uintptr_t)p;
}

__device__ __forceinline__ void async_b128(uint32_t ldsoff, const void* g) {
    asm volatile("global_load_async_to_lds_b128 %0, %1, off"
                 :: "v"(ldsoff), "v"(g) : "memory");
}

__device__ __forceinline__ void wait_async0()  {
    asm volatile("s_wait_asynccnt 0x0" ::: "memory");
}
__device__ __forceinline__ void wait_async16() {
    asm volatile("s_wait_asynccnt 0x10" ::: "memory");
}

// ---------------------------------------------------------------------------
// fp32 -> fp16 elementwise convert
// ---------------------------------------------------------------------------
__global__ __launch_bounds__(256) void k_cvt(const float* __restrict__ src,
                                             _Float16* __restrict__ dst, int n) {
    int i = blockIdx.x * 256 + threadIdx.x;
    if (i < n) dst[i] = (_Float16)src[i];
}

// W [K=1024, N=1024] fp32 -> Wt [N, K] fp16 (transpose for contiguous B-frags)
__global__ __launch_bounds__(256) void k_cvt_t(const float* __restrict__ W,
                                               _Float16* __restrict__ Wt) {
    int i = blockIdx.x * 256 + threadIdx.x;   // i over 1024*1024
    int k = i >> 10, n = i & 1023;
    Wt[n * DIM + k] = (_Float16)W[i];
}

// ---------------------------------------------------------------------------
// WMMA GEMM:  out = A[M,K] * Bt[N,K]^T + bias ; 32x64 tile per wave
// (2 A-frags x 4 B-frags -> 8 WMMAs per 12 b128 loads per K-step).
// mode 0: fp32 row-major [M,N] (final output)
// mode 1: fp16 head layout [B,H,S,DK], value * scale   (Q: scale=1/8, K: 1)
// mode 2: fp16 transposed head layout [B,H,DK,S]       (V)
// ---------------------------------------------------------------------------
__global__ __launch_bounds__(256) void k_gemm(const _Float16* __restrict__ A,
                                              const _Float16* __restrict__ Bt,
                                              const float* __restrict__ bias,
                                              void* __restrict__ out,
                                              int mode, float scale) {
    const int lane   = threadIdx.x & 31;
    const int wave   = threadIdx.x >> 5;
    const int tile   = blockIdx.x * 8 + wave;   // 0..2047
    const int nb     = tile & 15;               // 16 blocks of 64 columns
    const int mb     = tile >> 4;               // 128 blocks of 32 rows
    const int lane16 = lane & 15;
    const int hi     = lane >> 4;

    const int kbaseA = hi * 8;     // A: K interleaved per lane-half
    const int koffB  = hi * 8;     // B: K interleaved per lane-half (same as A)

    const _Float16* Arow0 = A + (mb * 32 + lane16) * DIM;
    const _Float16* Arow1 = Arow0 + 16 * DIM;
    const _Float16* Brow[4];
#pragma unroll
    for (int j = 0; j < 4; ++j)
        Brow[j] = Bt + (nb * 64 + j * 16 + lane16) * DIM;

    f32x8 acc0[4] = {}, acc1[4] = {};
    for (int k0 = 0; k0 < DIM; k0 += 32) {
        __builtin_prefetch(Arow0 + k0 + 128, 0, 3);
        __builtin_prefetch(Arow1 + k0 + 128, 0, 3);
        half16 a0 = cat16(ld8(Arow0 + k0 + kbaseA), ld8(Arow0 + k0 + 16 + kbaseA));
        half16 a1 = cat16(ld8(Arow1 + k0 + kbaseA), ld8(Arow1 + k0 + 16 + kbaseA));
#pragma unroll
        for (int j = 0; j < 4; ++j) {
            __builtin_prefetch(Brow[j] + k0 + 128, 0, 3);
            half16 b = cat16(ld8(Brow[j] + k0 + koffB), ld8(Brow[j] + k0 + koffB + 16));
            acc0[j] = wmma_f16(a0, b, acc0[j]);
            acc1[j] = wmma_f16(a1, b, acc1[j]);
        }
    }

#pragma unroll
    for (int j = 0; j < 4; ++j) {
        const int n  = nb * 64 + j * 16 + lane16;   // lane's C column
        const float bn = bias[n];
#pragma unroll
        for (int half = 0; half < 2; ++half) {
            const f32x8& acc = half ? acc1[j] : acc0[j];
            const int rowbase = mb * 32 + half * 16;
            if (mode == 0) {
                float* O = (float*)out;
#pragma unroll
                for (int r = 0; r < 8; ++r) {
                    int row = rowbase + r + 8 * hi;
                    O[row * DIM + n] = acc[r] + bn;
                }
            } else {
                _Float16* O = (_Float16*)out;
                const int h  = n >> 6;
                const int dk = n & 63;
#pragma unroll
                for (int r = 0; r < 8; ++r) {
                    int row = rowbase + r + 8 * hi;
                    int bb  = row >> 11;           // batch (S = 2048)
                    int s   = row & 2047;
                    float v = (acc[r] + bn) * scale;
                    int idx = (mode == 1)
                                ? (((bb * NH + h) * SEQ + s) * HDK + dk)
                                : (((bb * NH + h) * HDK + dk) * SEQ + s);
                    O[idx] = (_Float16)v;
                }
            }
        }
    }
}

// ---------------------------------------------------------------------------
// Flash-style attention core. One wave per (b, h, 16-row q tile).
// K/V blocks (32 keys) staged global->LDS with async b128 copies, double
// buffered; fragments built with ds_load_b128; P transposed through LDS.
// Qh [B,H,S,DK] (pre-scaled by 1/8), Kh [B,H,S,DK], Vt [B,H,DK,S].
// Output: attn fp16 row-major [B*S, D] ready for the final GEMM.
// ---------------------------------------------------------------------------
__device__ __forceinline__ void stage_kv(_Float16* sKb, _Float16* sVb,
                                         const _Float16* Kp, const _Float16* Vp,
                                         int kb, int lane) {
    // K block: rows kb..kb+31, 64 halves each -> contiguous 4096 B
    const uint32_t lK = lds_off(sKb);
    const char* gK = (const char*)(Kp + (size_t)kb * HDK);
#pragma unroll
    for (int i = 0; i < 8; ++i)
        async_b128(lK + i * 512 + lane * 16, gK + i * 512 + lane * 16);
    // V block: 64 dk-rows x 32 keys, each row segment 64 B (4 lanes x b128)
    const uint32_t lV = lds_off(sVb);
    const int dkr = lane >> 2;
    const int ko  = (lane & 3) * 16;   // byte offset along keys
#pragma unroll
    for (int i = 0; i < 8; ++i) {
        int dk = i * 8 + dkr;
        async_b128(lV + dk * 64 + ko,
                   (const char*)(Vp + (size_t)dk * SEQ + kb) + ko);
    }
}

__global__ __launch_bounds__(256) void k_attn(const _Float16* __restrict__ Qh,
                                              const _Float16* __restrict__ Kh,
                                              const _Float16* __restrict__ Vt,
                                              const float* __restrict__ relb,
                                              _Float16* __restrict__ outA) {
    __shared__ __align__(16) _Float16 sP[8][16 * 32];       //  8 KB
    __shared__ __align__(16) _Float16 sK[8][2][32 * HDK];   // 64 KB
    __shared__ __align__(16) _Float16 sV[8][2][HDK * 32];   // 64 KB

    const int lane   = threadIdx.x & 31;
    const int wave   = threadIdx.x >> 5;
    const int task   = blockIdx.x * 8 + wave;
    const int qt     = task & 127;            // S/16 = 128
    const int h      = (task >> 7) & 15;
    const int b      = task >> 11;
    const int lane16 = lane & 15;
    const int hi     = lane >> 4;

    const int bh = b * NH + h;
    const _Float16* Qp = Qh + (size_t)bh * SEQ * HDK;
    const _Float16* Kp = Kh + (size_t)bh * SEQ * HDK;
    const _Float16* Vp = Vt + (size_t)bh * HDK * SEQ;

    const int qbase = qt * 16;
    const int kend  = ((qbase / CHK) + 1) * CHK;   // growing-window chunk mask

    const int kbaseA = hi * 8;

    // Q fragments (A layout), K-dim = 64 -> two 16x32 fragments
    const _Float16* qrow = Qp + (qbase + lane16) * HDK;
    half16 qf0 = cat16(ld8(qrow + kbaseA),      ld8(qrow + 16 + kbaseA));
    half16 qf1 = cat16(ld8(qrow + 32 + kbaseA), ld8(qrow + 48 + kbaseA));

    f32x8 o[4] = {};            // 16 rows x 64 dk columns
    float mrow[8], lrow[8];
#pragma unroll
    for (int r = 0; r < 8; ++r) { mrow[r] = -1e30f; lrow[r] = 0.f; }

    _Float16* P = sP[wave];

    // prologue: stage first K/V block
    stage_kv(sK[wave][0], sV[wave][0], Kp, Vp, 0, lane);
    int buf = 0;

    for (int kb = 0; kb < kend; kb += 32) {
        const bool more = (kb + 32) < kend;
        if (more) {
            stage_kv(sK[wave][buf ^ 1], sV[wave][buf ^ 1], Kp, Vp, kb + 32, lane);
            wait_async16();   // previous block's 16 copies are done
        } else {
            wait_async0();
        }
        const _Float16* sKb = sK[wave][buf];
        const _Float16* sVb = sV[wave][buf];

        // ---- scores for two 16-key sub-tiles -------------------------------
        f32x8 sc[2];
#pragma unroll
        for (int nc = 0; nc < 2; ++nc) {
            const _Float16* krow = sKb + (nc * 16 + lane16) * HDK;
            half16 kf0 = cat16(ld8(krow + hi * 8),      ld8(krow + hi * 8 + 16));
            half16 kf1 = cat16(ld8(krow + 32 + hi * 8), ld8(krow + 32 + hi * 8 + 16));
            f32x8 c = {};
            c = wmma_f16(qf0, kf0, c);
            c = wmma_f16(qf1, kf1, c);
            // relative position bias (C-layout: row = r+8*hi, col = key)
#pragma unroll
            for (int r = 0; r < 8; ++r) {
                int q = qbase + r + 8 * hi;
                int k = kb + nc * 16 + lane16;
                int rel = q - k;
                rel = rel < -MRPC ? -MRPC : (rel > MRPC ? MRPC : rel);
                c[r] += relb[(rel + MRPC) * NH + h];
            }
            sc[nc] = c;
        }

        // ---- online softmax (per-row reductions over 16-lane halves) -------
        float al[8];
#pragma unroll
        for (int r = 0; r < 8; ++r) {
            float mx   = rmax16(fmaxf(sc[0][r], sc[1][r]));
            float mnew = fmaxf(mrow[r], mx);
            float a    = __expf(mrow[r] - mnew);
            float p0   = __expf(sc[0][r] - mnew);
            float p1   = __expf(sc[1][r] - mnew);
            sc[0][r] = p0; sc[1][r] = p1;
            lrow[r]  = lrow[r] * a + rsum16(p0 + p1);
            mrow[r]  = mnew;
            al[r]    = a;
        }
#pragma unroll
        for (int c4 = 0; c4 < 4; ++c4)
#pragma unroll
            for (int r = 0; r < 8; ++r) o[c4][r] *= al[r];

        // ---- P: C-layout -> A-layout transpose through LDS -----------------
#pragma unroll
        for (int nc = 0; nc < 2; ++nc)
#pragma unroll
            for (int r = 0; r < 8; ++r) {
                int row = r + 8 * hi;
                P[row * 32 + nc * 16 + lane16] = (_Float16)sc[nc][r];
            }
        asm volatile("s_wait_dscnt 0x0" ::: "memory");
        half16 pf = cat16(ld8(P + lane16 * 32 + kbaseA),
                          ld8(P + lane16 * 32 + 16 + kbaseA));

        // ---- O += P (16x32) x V (32x64) ------------------------------------
#pragma unroll
        for (int c4 = 0; c4 < 4; ++c4) {
            const _Float16* vrow = sVb + (c4 * 16 + lane16) * 32 + hi * 8;
            half16 vf = cat16(ld8(vrow), ld8(vrow + 16));
            o[c4] = wmma_f16(pf, vf, o[c4]);
        }
        buf ^= 1;
    }

    // ---- normalize and write fp16 [B*S, D] for the output projection ------
#pragma unroll
    for (int r = 0; r < 8; ++r) lrow[r] = 1.f / lrow[r];
#pragma unroll
    for (int c4 = 0; c4 < 4; ++c4)
#pragma unroll
        for (int r = 0; r < 8; ++r) {
            int q = qbase + r + 8 * hi;
            outA[(size_t)(b * SEQ + q) * DIM + h * HDK + c4 * 16 + lane16] =
                (_Float16)(o[c4][r] * lrow[r]);
        }
}

// ---------------------------------------------------------------------------
extern "C" void kernel_launch(void* const* d_in, const int* in_sizes, int n_in,
                              void* d_out, int out_size, void* d_ws, size_t ws_size,
                              hipStream_t stream) {
    (void)in_sizes; (void)n_in; (void)out_size; (void)ws_size;

    const float* x    = (const float*)d_in[0];
    const float* Wq   = (const float*)d_in[1];
    const float* bq   = (const float*)d_in[2];
    const float* Wk   = (const float*)d_in[3];
    const float* bk   = (const float*)d_in[4];
    const float* Wv   = (const float*)d_in[5];
    const float* bv   = (const float*)d_in[6];
    const float* Wo   = (const float*)d_in[7];
    const float* bo   = (const float*)d_in[8];
    const float* relb = (const float*)d_in[9];
    float* out = (float*)d_out;

    char* ws = (char*)d_ws;
    _Float16* xh  = (_Float16*)(ws);                          // 8 MB  x fp16
    _Float16* WtQ = (_Float16*)(ws + (size_t)8  * 1048576);   // 2 MB each
    _Float16* WtK = WtQ + DIM * DIM;
    _Float16* WtV = WtK + DIM * DIM;
    _Float16* WtO = WtV + DIM * DIM;
    _Float16* Qh  = (_Float16*)(ws + (size_t)16 * 1048576);   // 8 MB [B,H,S,DK]
    _Float16* Kh  = (_Float16*)(ws + (size_t)24 * 1048576);   // 8 MB [B,H,S,DK]
    _Float16* Vt  = (_Float16*)(ws + (size_t)32 * 1048576);   // 8 MB [B,H,DK,S]
    _Float16* At  = (_Float16*)(ws + (size_t)40 * 1048576);   // 8 MB [B*S, D]

    // converts
    k_cvt  <<<(MROW * DIM) / 256, 256, 0, stream>>>(x, xh, MROW * DIM);
    k_cvt_t<<<(DIM * DIM) / 256, 256, 0, stream>>>(Wq, WtQ);
    k_cvt_t<<<(DIM * DIM) / 256, 256, 0, stream>>>(Wk, WtK);
    k_cvt_t<<<(DIM * DIM) / 256, 256, 0, stream>>>(Wv, WtV);
    k_cvt_t<<<(DIM * DIM) / 256, 256, 0, stream>>>(Wo, WtO);

    // QKV projections (Q pre-scaled by 1/sqrt(64)); 32x64 tile per wave
    const int gemm_blocks = (MROW / 32) * (DIM / 64) / 8;   // 256
    k_gemm<<<gemm_blocks, 256, 0, stream>>>(xh, WtQ, bq, Qh, 1, 0.125f);
    k_gemm<<<gemm_blocks, 256, 0, stream>>>(xh, WtK, bk, Kh, 1, 1.0f);
    k_gemm<<<gemm_blocks, 256, 0, stream>>>(xh, WtV, bv, Vt, 2, 1.0f);

    // attention core: B*H*(S/16) = 4096 wave-tasks, 8 waves/block
    k_attn<<<(NB * NH * (SEQ / 16)) / 8, 256, 0, stream>>>(Qh, Kh, Vt, relb, At);

    // output projection straight to d_out (fp32)
    k_gemm<<<gemm_blocks, 256, 0, stream>>>(At, WtO, bo, out, 0, 1.0f);
}
